// GNN_topexpert_52948356825449
// MI455X (gfx1250) — compile-verified, matrix operand order checked by reading, not run on hardware
//
#include <hip/hip_runtime.h>
#include <stdint.h>

#define N_NODES   131072
#define N_EDGES   262144
#define N_GRAPHS  4096
#define EMB       300
#define EMB_P     320
#define HID       600
#define HID_P     640
#define N_LAYERS  5
#define N_EXPERTS 8
#define EXP_N     96
#define EXP_NP    128

typedef __attribute__((ext_vector_type(16))) __bf16 v16bf;
typedef __attribute__((ext_vector_type(8)))  float  v8f;
typedef __attribute__((ext_vector_type(4)))  int    v4i;

// ---------------------------------------------------------------------------
// WMMA GEMM with CDNA5 async global->LDS copies and double buffering.
//   C[M x Npad] = A[M x Kpad](bf16, row-major) @ B(bf16, PRE-TRANSPOSED
//   [Npad x Kpad], K-major) + bias
// 256 threads = 8 wave32, block tile 128x64, BK=32. Per K-step each wave runs
// a 2x2 grid of v_wmma_f32_16x16x32_bf16. Tiles are staged into LDS with
// global_load_async_to_lds_b128 (ASYNCcnt), double buffered: tile k+1 streams
// in while tile k is consumed; one s_wait_asynccnt + barrier per step.
// Fragment reads are contiguous ds_load_b128 matching the 16-bit operand
// layout (lanes 0-15: K{0..7,16..23}; lanes 16-31: K{8..15,24..31}).
// Requires M%128==0, Npad%64==0, Kpad%32==0 (buffers padded accordingly).
// ---------------------------------------------------------------------------
#define BM 128
#define BN 64
#define BK 32
#define SA_LD 40                       // padded LDS row stride (elems), 80B
#define SB_LD 40
#define SA_ELEMS (BM * SA_LD)          // 5120 elems
#define SB_ELEMS (BN * SB_LD)          // 2560 elems
#define BUF_ELEMS (SA_ELEMS + SB_ELEMS)
#define SA_BYTES (SA_ELEMS * 2)        // 10240
#define BUF_BYTES (BUF_ELEMS * 2)      // 15360
#define SMEM_BYTES (2 * BUF_BYTES)     // 30720

__device__ __forceinline__ void async_copy_b128(uint32_t lds_off, const __bf16* g)
{
    asm volatile("global_load_async_to_lds_b128 %0, %1, off"
                 :: "v"(lds_off), "v"((unsigned long long)(uintptr_t)g)
                 : "memory");
}

__device__ __forceinline__ void wait_async0()
{
    asm volatile("s_wait_asynccnt 0x0" ::: "memory");
}

__global__ __launch_bounds__(256)
void k_gemm_bf16(const __bf16* __restrict__ A, int lda,
                 const __bf16* __restrict__ Bt, int ldbT,   // [Npad x Kpad]
                 const float* __restrict__ bias, int nBias,
                 float* __restrict__ Cf, __bf16* __restrict__ Cbf, int ldc,
                 int Kpad, int doRelu)
{
    extern __shared__ __bf16 smem[];   // [2][sA 128x40 | sBt 64x40]

    const int tid  = threadIdx.x;
    const int lane = tid & 31;
    const int wave = tid >> 5;
    const int wm   = wave & 3;         // 4 waves over M (32 rows each)
    const int wn   = wave >> 2;        // 2 waves over N (32 cols each)

    const int rowBase = blockIdx.y * BM;
    const int colBase = blockIdx.x * BN;

    const int hsel = lane >> 4;        // K-half select
    const int lr   = lane & 15;
    const int kb   = hsel * 8;

    // per-thread async-copy coordinates (constant across K steps)
    const int ar0 = tid >> 2;          // A rows, pass 0: 0..63
    const int ar1 = (256 + tid) >> 2;  // A rows, pass 1: 64..127
    const int ac  = (tid & 3) * 8;     // 8-elem chunk within 32-wide K slab
    const int bcol = tid >> 2;         // B cols: 0..63

    const __bf16* Arow0 = A  + (size_t)(rowBase + ar0) * lda + ac;
    const __bf16* Arow1 = A  + (size_t)(rowBase + ar1) * lda + ac;
    const __bf16* Brow  = Bt + (size_t)(colBase + bcol) * ldbT + ac;

    const uint32_t aoff0 = (uint32_t)(ar0 * SA_LD + ac) * 2;
    const uint32_t aoff1 = (uint32_t)(ar1 * SA_LD + ac) * 2;
    const uint32_t boff  = (uint32_t)(SA_BYTES + (bcol * SB_LD + ac) * 2);

    auto issue = [&](int k0, int buf) {
        uint32_t base = (uint32_t)buf * BUF_BYTES;
        async_copy_b128(base + aoff0, Arow0 + k0);
        async_copy_b128(base + aoff1, Arow1 + k0);
        async_copy_b128(base + boff,  Brow  + k0);
    };

    v8f acc[2][2] = {};
    const int steps = Kpad / BK;

    issue(0, 0);
    wait_async0();
    __syncthreads();

    for (int s = 0; s < steps; ++s) {
        const int cur = s & 1;
        if (s + 1 < steps) issue((s + 1) * BK, 1 - cur);

        const __bf16* sA = smem + cur * BUF_ELEMS;
        const __bf16* sB = sA + SA_ELEMS;

        v16bf bfrag[2];
        #pragma unroll
        for (int sn = 0; sn < 2; ++sn) {
            const __bf16* bp = sB + (wn * 32 + sn * 16 + lr) * SB_LD;
            ((v4i*)&bfrag[sn])[0] = *(const v4i*)(bp + kb);
            ((v4i*)&bfrag[sn])[1] = *(const v4i*)(bp + kb + 16);
        }
        #pragma unroll
        for (int sm = 0; sm < 2; ++sm) {
            v16bf afrag;
            const __bf16* ap = sA + (wm * 32 + sm * 16 + lr) * SA_LD;
            ((v4i*)&afrag)[0] = *(const v4i*)(ap + kb);
            ((v4i*)&afrag)[1] = *(const v4i*)(ap + kb + 16);
            #pragma unroll
            for (int sn = 0; sn < 2; ++sn) {
                acc[sm][sn] = __builtin_amdgcn_wmma_f32_16x16x32_bf16(
                    false, afrag, false, bfrag[sn],
                    (short)0, acc[sm][sn], false, false);
            }
        }

        wait_async0();
        __syncthreads();
    }

    #pragma unroll
    for (int sm = 0; sm < 2; ++sm) {
        #pragma unroll
        for (int sn = 0; sn < 2; ++sn) {
            int colg = colBase + wn * 32 + sn * 16 + lr;
            float bi = (bias && colg < nBias) ? bias[colg] : 0.0f;
            #pragma unroll
            for (int v = 0; v < 8; ++v) {
                int rowg = rowBase + wm * 32 + sm * 16 + v + hsel * 8;
                float o = acc[sm][sn][v] + bi;
                if (doRelu) o = fmaxf(o, 0.0f);
                size_t idx = (size_t)rowg * ldc + colg;
                if (Cf)  Cf[idx]  = o;
                if (Cbf) Cbf[idx] = (__bf16)o;
            }
        }
    }
}

// ---------------------------------------------------------------------------
// Helper kernels
// ---------------------------------------------------------------------------
__global__ void k_init_h(const int* __restrict__ x, const float* __restrict__ ae1,
                         const float* __restrict__ ae2, float* __restrict__ h)
{
    int n = blockIdx.x, c = threadIdx.x;
    float v = 0.0f;
    if (c < EMB) {
        int a0 = x[n * 2 + 0], a1 = x[n * 2 + 1];
        v = ae1[(size_t)a0 * EMB + c] + ae2[(size_t)a1 * EMB + c];
    }
    h[(size_t)n * EMB_P + c] = v;
}

// pad + convert f32 [K x N] -> bf16 TRANSPOSED [Npad x Kpad] (K-major)
__global__ void k_pad_weight_t(const float* __restrict__ W, int K, int N,
                               __bf16* __restrict__ out, int Kpad, int Npad)
{
    int i = blockIdx.x * blockDim.x + threadIdx.x;
    if (i >= Kpad * Npad) return;
    int n = i / Kpad, k = i % Kpad;
    float v = (k < K && n < N) ? W[(size_t)k * N + n] : 0.0f;
    out[i] = (__bf16)v;
}

__global__ void k_agg_init(const float* __restrict__ h, const float* __restrict__ ee1l,
                           const float* __restrict__ ee2l, float* __restrict__ agg)
{
    int n = blockIdx.x, c = threadIdx.x;
    float v = 0.0f;
    if (c < EMB) v = h[(size_t)n * EMB_P + c] + ee1l[4 * EMB + c] + ee2l[c];
    agg[(size_t)n * EMB_P + c] = v;
}

__global__ void k_edge_scatter(const int* __restrict__ ei, const int* __restrict__ ea,
                               const float* __restrict__ h, const float* __restrict__ ee1l,
                               const float* __restrict__ ee2l, float* __restrict__ agg)
{
    int e = blockIdx.x, c = threadIdx.x;
    if (c >= EMB) return;
    int src = ei[e];
    int dst = ei[N_EDGES + e];
    int b0  = ea[e * 2 + 0];
    int b1  = ea[e * 2 + 1];
    float v = h[(size_t)src * EMB_P + c]
            + ee1l[(size_t)b0 * EMB + c]
            + ee2l[(size_t)b1 * EMB + c];
    atomicAdd(&agg[(size_t)dst * EMB_P + c], v);
}

__global__ void k_cvt_bf16(const float* __restrict__ in, __bf16* __restrict__ out,
                           size_t n)
{
    size_t i = (size_t)blockIdx.x * blockDim.x + threadIdx.x;
    if (i < n) out[i] = (__bf16)in[i];
}

__global__ void k_bn_stats(const float* __restrict__ X, float* __restrict__ stats,
                           int M, int ld, int N, int rowsPerBlock)
{
    int c = threadIdx.x;
    if (c >= N) return;
    int r0 = blockIdx.x * rowsPerBlock;
    int r1 = min(r0 + rowsPerBlock, M);
    float s = 0.0f, ss = 0.0f;
    for (int r = r0; r < r1; ++r) {
        float v = X[(size_t)r * ld + c];
        s += v; ss += v * v;
    }
    atomicAdd(&stats[c], s);
    atomicAdd(&stats[ld + c], ss);
}

__global__ void k_bn_apply(const float* __restrict__ X, const float* __restrict__ stats,
                           const float* __restrict__ g, const float* __restrict__ b,
                           float* __restrict__ outF, __bf16* __restrict__ outB,
                           int M, int ld, int N, int doRelu)
{
    int c = threadIdx.x;
    size_t r = blockIdx.x;
    float o = 0.0f;
    if (c < N) {
        float inv_m = 1.0f / (float)M;
        float mu  = stats[c] * inv_m;
        float var = stats[ld + c] * inv_m - mu * mu;
        float inv = rsqrtf(var + 1e-5f);
        o = g[c] * (X[r * ld + c] - mu) * inv + b[c];
        if (doRelu) o = fmaxf(o, 0.0f);
    }
    if (outF) outF[r * ld + c] = o;
    if (outB) outB[r * ld + c] = (__bf16)o;
}

__global__ void k_pool(const float* __restrict__ h, const int* __restrict__ batch,
                       float* __restrict__ sums, float* __restrict__ counts)
{
    int n = blockIdx.x, c = threadIdx.x;
    int g = batch[n];
    atomicAdd(&sums[(size_t)g * EMB_P + c], h[(size_t)n * EMB_P + c]);
    if (c == 0) atomicAdd(&counts[g], 1.0f);
}

__global__ void k_graph_vecs(const float* __restrict__ sums, const float* __restrict__ counts,
                             __bf16* __restrict__ sums_bf, __bf16* __restrict__ gout_bf)
{
    int g = blockIdx.x, c = threadIdx.x;
    size_t idx = (size_t)g * EMB_P + c;
    float s = (c < EMB) ? sums[idx] : 0.0f;
    float cnt = fmaxf(counts[g], 1.0f);
    sums_bf[idx] = (__bf16)s;
    gout_bf[idx] = (__bf16)(s / cnt);
}

__global__ void k_copy_cols(const float* __restrict__ src, int lds_, int ncols,
                            float* __restrict__ dst, int M)
{
    int i = blockIdx.x * blockDim.x + threadIdx.x;
    if (i >= M * ncols) return;
    int r = i / ncols, c = i % ncols;
    dst[i] = src[(size_t)r * lds_ + c];
}

__global__ void k_norm_cluster(const float* __restrict__ cluster, float* __restrict__ cn)
{
    __shared__ float ssum;
    int e = blockIdx.x, c = threadIdx.x;
    if (c == 0) ssum = 0.0f;
    __syncthreads();
    float v = (c < EMB) ? cluster[(size_t)e * EMB + c] : 0.0f;
    atomicAdd(&ssum, v * v);
    __syncthreads();
    float inv = 1.0f / fmaxf(sqrtf(ssum), 1e-6f);
    if (c < EMB) cn[(size_t)e * EMB + c] = v * inv;
}

__global__ __launch_bounds__(256)
void k_finalize(const float* __restrict__ z, const float* __restrict__ cn,
                float* __restrict__ zOut, float* __restrict__ qOut, int nGraphs)
{
    int lane = threadIdx.x & 31;
    int g = blockIdx.x * 8 + (threadIdx.x >> 5);
    if (g >= nGraphs) return;
    const float* zr = z + (size_t)g * EMB_P;

    float ss = 0.0f;
    for (int c = lane; c < EMB; c += 32) {
        float v = zr[c];
        zOut[(size_t)g * EMB + c] = v;
        ss += v * v;
    }
    #pragma unroll
    for (int m = 16; m > 0; m >>= 1) ss += __shfl_xor(ss, m, 32);
    float inv = 1.0f / fmaxf(sqrtf(ss), 1e-6f);

    float logit[N_EXPERTS];
    #pragma unroll
    for (int e = 0; e < N_EXPERTS; ++e) {
        const float* ce = cn + (size_t)e * EMB;
        float d = 0.0f;
        for (int c = lane; c < EMB; c += 32) d += zr[c] * ce[c];
        #pragma unroll
        for (int m = 16; m > 0; m >>= 1) d += __shfl_xor(d, m, 32);
        logit[e] = 10.0f * d * inv;
    }
    float mx = logit[0];
    #pragma unroll
    for (int e = 1; e < N_EXPERTS; ++e) mx = fmaxf(mx, logit[e]);
    float se = 0.0f;
    #pragma unroll
    for (int e = 0; e < N_EXPERTS; ++e) { logit[e] = __expf(logit[e] - mx); se += logit[e]; }
    if (lane < N_EXPERTS) qOut[(size_t)g * N_EXPERTS + lane] = logit[lane] / se;
}

static inline void launch_gemm(hipStream_t s,
                               const __bf16* A, int lda, const __bf16* Bt,
                               const float* bias, int nBias,
                               float* Cf, __bf16* Cbf, int ldc,
                               int M, int Npad, int Kpad, int relu)
{
    dim3 grid(Npad / BN, M / BM);
    k_gemm_bf16<<<grid, 256, SMEM_BYTES, s>>>(A, lda, Bt, Kpad, bias, nBias,
                                              Cf, Cbf, ldc, Kpad, relu);
}

extern "C" void kernel_launch(void* const* d_in, const int* in_sizes, int n_in,
                              void* d_out, int out_size, void* d_ws, size_t ws_size,
                              hipStream_t stream)
{
    const int*   x        = (const int*)  d_in[0];
    const int*   ei       = (const int*)  d_in[1];
    const int*   ea       = (const int*)  d_in[2];
    const int*   batch    = (const int*)  d_in[3];
    const float* atom1    = (const float*)d_in[4];
    const float* atom2    = (const float*)d_in[5];
    const float* eemb1    = (const float*)d_in[6];
    const float* eemb2    = (const float*)d_in[7];
    const float* mlp_w1   = (const float*)d_in[8];
    const float* mlp_b1   = (const float*)d_in[9];
    const float* mlp_w2   = (const float*)d_in[10];
    const float* mlp_b2   = (const float*)d_in[11];
    const float* bn_g     = (const float*)d_in[12];
    const float* bn_b     = (const float*)d_in[13];
    const float* gate_w1  = (const float*)d_in[14];
    const float* gate_b1  = (const float*)d_in[15];
    const float* gate_bng = (const float*)d_in[16];
    const float* gate_bnb = (const float*)d_in[17];
    const float* gate_w2  = (const float*)d_in[18];
    const float* gate_b2  = (const float*)d_in[19];
    const float* cluster  = (const float*)d_in[20];
    const float* exp_w    = (const float*)d_in[21];
    const float* exp_b    = (const float*)d_in[22];

    uint8_t* wsp = (uint8_t*)d_ws;
    auto alloc = [&](size_t bytes) -> void* {
        void* p = (void*)wsp;
        wsp += (bytes + 255) & ~(size_t)255;
        return p;
    };
    float*  h      = (float*) alloc((size_t)N_NODES * EMB_P * 4);
    float*  aggf   = (float*) alloc((size_t)N_NODES * EMB_P * 4);   // reused as pre-BN
    __bf16* aggbf  = (__bf16*)alloc((size_t)N_NODES * EMB_P * 2);
    __bf16* hidbf  = (__bf16*)alloc((size_t)N_NODES * HID_P * 2);
    __bf16* w1bf   = (__bf16*)alloc((size_t)N_LAYERS * EMB_P * HID_P * 2);  // [640x320] each
    __bf16* w2bf   = (__bf16*)alloc((size_t)N_LAYERS * HID_P * EMB_P * 2);  // [320x640] each
    __bf16* gw1bf  = (__bf16*)alloc((size_t)EMB_P * EMB_P * 2);
    __bf16* gw2bf  = (__bf16*)alloc((size_t)EMB_P * EMB_P * 2);
    __bf16* ewbf   = (__bf16*)alloc((size_t)EXP_NP * EMB_P * 2);
    float*  stats  = (float*) alloc((size_t)2 * EMB_P * 4);
    float*  sums   = (float*) alloc((size_t)N_GRAPHS * EMB_P * 4);
    float*  counts = (float*) alloc((size_t)N_GRAPHS * 4);
    __bf16* sumsbf = (__bf16*)alloc((size_t)N_GRAPHS * EMB_P * 2);
    __bf16* goutbf = (__bf16*)alloc((size_t)N_GRAPHS * EMB_P * 2);
    float*  zh     = (float*) alloc((size_t)N_GRAPHS * EMB_P * 4);
    __bf16* zhbf   = (__bf16*)alloc((size_t)N_GRAPHS * EMB_P * 2);
    float*  zf32   = (float*) alloc((size_t)N_GRAPHS * EMB_P * 4);
    float*  clfpad = (float*) alloc((size_t)N_GRAPHS * EXP_NP * 4);
    float*  cn     = (float*) alloc((size_t)N_EXPERTS * EMB * 4);

    float* out_clf = (float*)d_out;
    float* out_z   = out_clf + (size_t)N_GRAPHS * EXP_N;
    float* out_q   = out_z   + (size_t)N_GRAPHS * EMB;

    // weight conversion -> padded, pre-transposed bf16 ([Npad x Kpad], K-major)
    {
        int n1 = EMB_P * HID_P;
        for (int l = 0; l < N_LAYERS; ++l) {
            k_pad_weight_t<<<(n1 + 255) / 256, 256, 0, stream>>>(
                mlp_w1 + (size_t)l * EMB * HID, EMB, HID, w1bf + (size_t)l * n1, EMB_P, HID_P);
            k_pad_weight_t<<<(n1 + 255) / 256, 256, 0, stream>>>(
                mlp_w2 + (size_t)l * HID * EMB, HID, EMB, w2bf + (size_t)l * n1, HID_P, EMB_P);
        }
        int n2 = EMB_P * EMB_P;
        k_pad_weight_t<<<(n2 + 255) / 256, 256, 0, stream>>>(gate_w1, EMB, EMB, gw1bf, EMB_P, EMB_P);
        k_pad_weight_t<<<(n2 + 255) / 256, 256, 0, stream>>>(gate_w2, EMB, EMB, gw2bf, EMB_P, EMB_P);
        int n3 = EMB_P * EXP_NP;
        k_pad_weight_t<<<(n3 + 255) / 256, 256, 0, stream>>>(exp_w, EMB, EXP_N, ewbf, EMB_P, EXP_NP);
    }

    k_init_h<<<N_NODES, EMB_P, 0, stream>>>(x, atom1, atom2, h);

    const size_t nElemNode = (size_t)N_NODES * EMB_P;
    for (int l = 0; l < N_LAYERS; ++l) {
        const float* ee1l = eemb1 + (size_t)l * 6 * EMB;
        const float* ee2l = eemb2 + (size_t)l * 3 * EMB;

        k_agg_init<<<N_NODES, EMB_P, 0, stream>>>(h, ee1l, ee2l, aggf);
        k_edge_scatter<<<N_EDGES, EMB_P, 0, stream>>>(ei, ea, h, ee1l, ee2l, aggf);
        k_cvt_bf16<<<(int)((nElemNode + 255) / 256), 256, 0, stream>>>(aggf, aggbf, nElemNode);

        launch_gemm(stream, aggbf, EMB_P, w1bf + (size_t)l * EMB_P * HID_P,
                    mlp_b1 + (size_t)l * HID, HID, nullptr, hidbf, HID_P,
                    N_NODES, HID_P, EMB_P, 1);
        launch_gemm(stream, hidbf, HID_P, w2bf + (size_t)l * HID_P * EMB_P,
                    mlp_b2 + (size_t)l * EMB, EMB, aggf, nullptr, EMB_P,
                    N_NODES, EMB_P, HID_P, 0);

        hipMemsetAsync(stats, 0, 2 * EMB_P * sizeof(float), stream);
        k_bn_stats<<<N_NODES / 512, EMB_P, 0, stream>>>(aggf, stats, N_NODES, EMB_P, EMB, 512);
        k_bn_apply<<<N_NODES, EMB_P, 0, stream>>>(aggf, stats,
            bn_g + (size_t)l * EMB, bn_b + (size_t)l * EMB,
            h, nullptr, N_NODES, EMB_P, EMB, (l < N_LAYERS - 1) ? 1 : 0);
    }

    hipMemsetAsync(sums, 0, (size_t)N_GRAPHS * EMB_P * sizeof(float), stream);
    hipMemsetAsync(counts, 0, (size_t)N_GRAPHS * sizeof(float), stream);
    k_pool<<<N_NODES, EMB_P, 0, stream>>>(h, batch, sums, counts);
    k_graph_vecs<<<N_GRAPHS, EMB_P, 0, stream>>>(sums, counts, sumsbf, goutbf);

    launch_gemm(stream, sumsbf, EMB_P, gw1bf, gate_b1, EMB,
                zh, nullptr, EMB_P, N_GRAPHS, EMB_P, EMB_P, 0);
    hipMemsetAsync(stats, 0, 2 * EMB_P * sizeof(float), stream);
    k_bn_stats<<<N_GRAPHS / 512, EMB_P, 0, stream>>>(zh, stats, N_GRAPHS, EMB_P, EMB, 512);
    k_bn_apply<<<N_GRAPHS, EMB_P, 0, stream>>>(zh, stats, gate_bng, gate_bnb,
                                               nullptr, zhbf, N_GRAPHS, EMB_P, EMB, 1);
    launch_gemm(stream, zhbf, EMB_P, gw2bf, gate_b2, EMB,
                zf32, nullptr, EMB_P, N_GRAPHS, EMB_P, EMB_P, 0);

    launch_gemm(stream, goutbf, EMB_P, ewbf, exp_b, EXP_N,
                clfpad, nullptr, EXP_NP, N_GRAPHS, EXP_NP, EMB_P, 0);
    k_copy_cols<<<(N_GRAPHS * EXP_N + 255) / 256, 256, 0, stream>>>(
        clfpad, EXP_NP, EXP_N, out_clf, N_GRAPHS);

    k_norm_cluster<<<N_EXPERTS, EMB_P, 0, stream>>>(cluster, cn);
    k_finalize<<<N_GRAPHS / 8, 256, 0, stream>>>(zf32, cn, out_z, out_q, N_GRAPHS);
}
